// MambaMixer_19104014532979
// MI455X (gfx1250) — matrix operand order repaired and checked by reading, not compiled, not run on hardware
//
#include <hip/hip_runtime.h>
#include <hip/hip_bf16.h>

#define D_MODEL 2048
#define D_STATE 16
#define D_CONV  4
#define D_INNER 4096
#define DT_RANK 128
#define BATCH   2
#define SEQ     1024
#define NTOK    (BATCH * SEQ)            // 2048 tokens
#define XDBL_LD (DT_RANK + 2 * D_STATE)  // 160

typedef __attribute__((ext_vector_type(16))) __bf16         v16bf;
typedef __attribute__((ext_vector_type(16))) unsigned short v16us;
typedef __attribute__((ext_vector_type(8)))  unsigned short v8us;
typedef __attribute__((ext_vector_type(8)))  float          v8f;

// ---------------------------------------------------------------------------
// fp32 -> bf16 round-to-nearest-even (bit trick)
// ---------------------------------------------------------------------------
__device__ __forceinline__ unsigned short f2bf(float f) {
  unsigned int u = __float_as_uint(f);
  u += 0x7FFFu + ((u >> 16) & 1u);
  return (unsigned short)(u >> 16);
}

// ---------------------------------------------------------------------------
// Bulk fp32 -> packed bf16 conversion; 8 elements per thread.
// ---------------------------------------------------------------------------
__global__ __launch_bounds__(256)
void f32_to_bf16_kernel(const float* __restrict__ in,
                        unsigned short* __restrict__ out, int n8) {
  const int i = blockIdx.x * blockDim.x + threadIdx.x;
  if (i >= n8) return;
  const float4* p = (const float4*)in + (size_t)i * 2;
  float4 a = p[0], b = p[1];
  v8us o;
  o[0] = f2bf(a.x); o[1] = f2bf(a.y); o[2] = f2bf(a.z); o[3] = f2bf(a.w);
  o[4] = f2bf(b.x); o[5] = f2bf(b.y); o[6] = f2bf(b.z); o[7] = f2bf(b.w);
  ((v8us*)out)[i] = o;
}

// ---------------------------------------------------------------------------
// A-fragment: 16x32 (MxK) bf16, ISA 7.12.2 layout, from packed bf16.
// lane<16  -> row = lane,    K = {0..7, 16..23}
// lane>=16 -> row = lane-16, K = {8..15, 24..31}
// ---------------------------------------------------------------------------
__device__ __forceinline__ v16bf load_frag_a(const unsigned short* p, int ld) {
  const int lane = threadIdx.x & 31;
  const unsigned short* rp = p + (size_t)(lane & 15) * ld + ((lane & 16) ? 8 : 0);
  union { v16us v; struct { v8us lo, hi; } s; } u;
  u.s.lo = *(const v8us*)(rp);        // K kb .. kb+7
  u.s.hi = *(const v8us*)(rp + 16);   // K kb+16 .. kb+23
  __builtin_prefetch(rp + 64, 0, 3);  // k-tile after next -> global_prefetch_b8
  return __builtin_bit_cast(v16bf, u.v);
}

// ---------------------------------------------------------------------------
// B-fragment: 32x16 (KxN) bf16; math-B column n = row n of (N x K) weights.
// lane<16 -> col = lane, K = {0..7, 16..23} ; lane>=16 -> col = lane-16, K = {8..15, 24..31}
// ---------------------------------------------------------------------------
__device__ __forceinline__ v16bf load_frag_b(const unsigned short* p, int ld) {
  const int lane = threadIdx.x & 31;
  const unsigned short* rp = p + (size_t)(lane & 15) * ld + ((lane & 16) ? 8 : 0);
  union { v16us v; struct { v8us lo, hi; } s; } u;
  u.s.lo = *(const v8us*)(rp);
  u.s.hi = *(const v8us*)(rp + 16);
  __builtin_prefetch(rp + 64, 0, 3);
  return __builtin_bit_cast(v16bf, u.v);
}

// ---------------------------------------------------------------------------
// C = A(MxK) * B^T (B stored N x K, both packed bf16), f32 accumulate.
// Wave: (WMT*16) x (WNT*16); block = BWM x BWN waves.
// 2-stage software pipeline over K (double-buffered fragments, k-unroll 64).
// Requires K % 64 == 0.
// EPI: 0 = plain f32 store
//      1 = softplus(acc + bias[col])          (delta)
//      2 = f32 store + bf16 aux store to Cbf  (x_dbl)
// ---------------------------------------------------------------------------
template <int WMT, int WNT, int BWM, int BWN, int EPI>
__global__ __launch_bounds__(32 * BWM * BWN)
void gemm_wmma_bf16(const unsigned short* __restrict__ Ap, int lda,
                    const unsigned short* __restrict__ Bp, int ldb,
                    float* __restrict__ Cp, int ldc, int K,
                    const float* __restrict__ bias,
                    unsigned short* __restrict__ Cbf) {
  const int wave  = threadIdx.x >> 5;
  const int lane  = threadIdx.x & 31;
  const int waveM = wave / BWN;
  const int waveN = wave % BWN;
  const int row0  = blockIdx.x * (BWM * WMT * 16) + waveM * (WMT * 16);
  const int col0  = blockIdx.y * (BWN * WNT * 16) + waveN * (WNT * 16);

  const unsigned short* Arow[WMT];
  const unsigned short* Brow[WNT];
#pragma unroll
  for (int i = 0; i < WMT; ++i) Arow[i] = Ap + (size_t)(row0 + i * 16) * lda;
#pragma unroll
  for (int j = 0; j < WNT; ++j) Brow[j] = Bp + (size_t)(col0 + j * 16) * ldb;

  v8f acc[WMT][WNT];
#pragma unroll
  for (int i = 0; i < WMT; ++i)
#pragma unroll
    for (int j = 0; j < WNT; ++j) acc[i][j] = (v8f)0.0f;

  v16bf af0[WMT], bf0[WNT], af1[WMT], bf1[WNT];

  // prologue: stage k = 0
#pragma unroll
  for (int i = 0; i < WMT; ++i) af0[i] = load_frag_a(Arow[i], lda);
#pragma unroll
  for (int j = 0; j < WNT; ++j) bf0[j] = load_frag_b(Brow[j], ldb);

  for (int k0 = 0; k0 < K; k0 += 64) {
    // stage k0+32 while computing k0
#pragma unroll
    for (int i = 0; i < WMT; ++i) af1[i] = load_frag_a(Arow[i] + k0 + 32, lda);
#pragma unroll
    for (int j = 0; j < WNT; ++j) bf1[j] = load_frag_b(Brow[j] + k0 + 32, ldb);
#pragma unroll
    for (int i = 0; i < WMT; ++i)
#pragma unroll
      for (int j = 0; j < WNT; ++j)
        acc[i][j] = __builtin_amdgcn_wmma_f32_16x16x32_bf16(
            false, af0[i], false, bf0[j], (short)0, acc[i][j], false, false);

    // stage k0+64 (unless last) while computing k0+32
    if (k0 + 64 < K) {
#pragma unroll
      for (int i = 0; i < WMT; ++i) af0[i] = load_frag_a(Arow[i] + k0 + 64, lda);
#pragma unroll
      for (int j = 0; j < WNT; ++j) bf0[j] = load_frag_b(Brow[j] + k0 + 64, ldb);
    }
#pragma unroll
    for (int i = 0; i < WMT; ++i)
#pragma unroll
      for (int j = 0; j < WNT; ++j)
        acc[i][j] = __builtin_amdgcn_wmma_f32_16x16x32_bf16(
            false, af1[i], false, bf1[j], (short)0, acc[i][j], false, false);
  }

  // C/D layout: VGPR v -> M = v (+8 for upper half-wave); N = lane & 15
  const int rsub = (lane & 16) ? 8 : 0;
  const int csub = lane & 15;
#pragma unroll
  for (int i = 0; i < WMT; ++i)
#pragma unroll
    for (int j = 0; j < WNT; ++j)
#pragma unroll
      for (int v = 0; v < 8; ++v) {
        const int r = row0 + i * 16 + v + rsub;
        const int c = col0 + j * 16 + csub;
        float val = acc[i][j][v];
        if (EPI == 1) {  // delta = softplus(dt + b_dt)
          val += bias[c];
          val = (val > 20.0f) ? val : __logf(1.0f + __expf(val));
        }
        Cp[(size_t)r * ldc + c] = val;
        if (EPI == 2) Cbf[(size_t)r * ldc + c] = f2bf(val);
      }
}

// ---------------------------------------------------------------------------
// Causal depthwise conv1d (k=4, left pad 3) + bias + SiLU.
// x lives in the first D_INNER columns of xz (row stride 2*D_INNER).
// Writes fp32 (for the scan) and packed bf16 (for the x_dbl GEMM).
// ---------------------------------------------------------------------------
__global__ __launch_bounds__(256)
void conv_silu_kernel(const float* __restrict__ xz, const float* __restrict__ w,
                      const float* __restrict__ bias, float* __restrict__ xc,
                      unsigned short* __restrict__ xcb) {
  const int idx = blockIdx.x * blockDim.x + threadIdx.x;  // (b*SEQ + t)*D_INNER + d
  const int d  = idx & (D_INNER - 1);
  const int bt = idx >> 12;            // D_INNER == 2^12
  const int t  = bt & (SEQ - 1);
  const int b  = bt >> 10;             // SEQ == 2^10
  const float* xcol = xz + (size_t)b * SEQ * (2 * D_INNER) + d;
  float acc = bias[d];
#pragma unroll
  for (int k = 0; k < D_CONV; ++k) {
    const int tt = t - (D_CONV - 1) + k;
    if (tt >= 0) acc = fmaf(w[d * D_CONV + k], xcol[(size_t)tt * (2 * D_INNER)], acc);
  }
  const float s = acc / (1.0f + __expf(-acc));  // SiLU
  xc[idx]  = s;
  xcb[idx] = f2bf(s);
}

// ---------------------------------------------------------------------------
// Selective scan: one lane per (b, d) channel; 16-state in registers.
// B_t / C_t (32 floats per (b,t)) loaded one-per-lane and broadcast with
// wave32 shuffles (ds_bpermute). Fuses D-skip and SiLU(z) gating.
// Output written directly as bf16 (only consumer is the out-proj WMMA GEMM).
// ---------------------------------------------------------------------------
__global__ __launch_bounds__(256)
void scan_kernel(const float* __restrict__ delta, const float* __restrict__ u,
                 const float* __restrict__ xdbl, const float* __restrict__ xz,
                 const float* __restrict__ A_log, const float* __restrict__ Dsk,
                 unsigned short* __restrict__ ybf) {
  const int gid  = blockIdx.x * blockDim.x + threadIdx.x;  // 0 .. BATCH*D_INNER-1
  const int d    = gid & (D_INNER - 1);
  const int b    = gid >> 12;
  const int lane = threadIdx.x & 31;

  float Av[D_STATE], h[D_STATE];
#pragma unroll
  for (int n = 0; n < D_STATE; ++n) {
    Av[n] = -__expf(A_log[d * D_STATE + n]);
    h[n]  = 0.0f;
  }
  const float Dd = Dsk[d];

  const float*    dp  = delta + (size_t)b * SEQ * D_INNER + d;
  const float*    up  = u     + (size_t)b * SEQ * D_INNER + d;
  const float*    zp  = xz    + (size_t)b * SEQ * (2 * D_INNER) + D_INNER + d;
  const float*    bcp = xdbl  + (size_t)b * SEQ * XDBL_LD + DT_RANK + lane;
  unsigned short* yp  = ybf   + (size_t)b * SEQ * D_INNER + d;

  for (int t = 0; t < SEQ; ++t) {
    const float dt = dp[(size_t)t * D_INNER];
    const float ut = up[(size_t)t * D_INNER];
    const float zt = zp[(size_t)t * (2 * D_INNER)];
    const float bc = bcp[(size_t)t * XDBL_LD];  // lane n: B_t[n] (n<16) / C_t[n-16]
    const float du = dt * ut;
    float yv = 0.0f;
#pragma unroll
    for (int n = 0; n < D_STATE; ++n) {
      const float Bn = __shfl(bc, n, 32);
      const float Cn = __shfl(bc, n + D_STATE, 32);
      h[n] = __expf(dt * Av[n]) * h[n] + du * Bn;
      yv   = fmaf(h[n], Cn, yv);
    }
    yv = fmaf(Dd, ut, yv);                  // + D_skip * u
    yv *= zt / (1.0f + __expf(-zt));        // * SiLU(z)
    yp[(size_t)t * D_INNER] = f2bf(yv);
  }
}

// ---------------------------------------------------------------------------
extern "C" void kernel_launch(void* const* d_in, const int* in_sizes, int n_in,
                              void* d_out, int out_size, void* d_ws, size_t ws_size,
                              hipStream_t stream) {
  const float* hs     = (const float*)d_in[0];
  const float* W_in   = (const float*)d_in[1];
  const float* conv_w = (const float*)d_in[2];
  const float* conv_b = (const float*)d_in[3];
  const float* W_x    = (const float*)d_in[4];
  const float* W_dt   = (const float*)d_in[5];
  const float* b_dt   = (const float*)d_in[6];
  const float* A_log  = (const float*)d_in[7];
  const float* D_skip = (const float*)d_in[8];
  const float* W_out  = (const float*)d_in[9];
  float* out = (float*)d_out;

  // ---- workspace layout ---------------------------------------------------
  // fp32 region
  float* ws    = (float*)d_ws;
  float* xz    = ws;                                   // NTOK * 8192   (x | z)
  float* xconv = xz + (size_t)NTOK * 2 * D_INNER;      // NTOK * 4096
  float* xdbl  = xconv + (size_t)NTOK * D_INNER;       // NTOK * 160
  float* delta = xdbl + (size_t)NTOK * XDBL_LD;        // NTOK * 4096
  // packed-bf16 region (all sizes multiples of 8 -> 16B alignment preserved)
  unsigned short* bfbase   = (unsigned short*)(delta + (size_t)NTOK * D_INNER);
  unsigned short* hs_bf    = bfbase;                                    // 2048*2048
  unsigned short* W_in_bf  = hs_bf    + (size_t)NTOK * D_MODEL;         // 8192*2048
  unsigned short* W_x_bf   = W_in_bf  + (size_t)2 * D_INNER * D_MODEL;  // 160*4096
  unsigned short* W_dt_bf  = W_x_bf   + (size_t)XDBL_LD * D_INNER;      // 4096*128
  unsigned short* W_out_bf = W_dt_bf  + (size_t)D_INNER * DT_RANK;      // 2048*4096
  unsigned short* xconv_bf = W_out_bf + (size_t)D_MODEL * D_INNER;      // 2048*4096 (reused as y_bf)
  unsigned short* xdbl_bf  = xconv_bf + (size_t)NTOK * D_INNER;         // 2048*160

  const dim3 blk(256);
  auto cvt = [&](const float* src, unsigned short* dst, size_t n) {
    const int n8 = (int)(n / 8);
    f32_to_bf16_kernel<<<(n8 + 255) / 256, blk, 0, stream>>>(src, dst, n8);
  };

  // 0) stage weights + input as packed bf16 (convert once, multiply many)
  cvt(hs,    hs_bf,    (size_t)NTOK * D_MODEL);
  cvt(W_in,  W_in_bf,  (size_t)2 * D_INNER * D_MODEL);
  cvt(W_x,   W_x_bf,   (size_t)XDBL_LD * D_INNER);
  cvt(W_dt,  W_dt_bf,  (size_t)D_INNER * DT_RANK);
  cvt(W_out, W_out_bf, (size_t)D_MODEL * D_INNER);

  // 1) xz = hs @ W_in^T           M=2048 N=8192 K=2048
  gemm_wmma_bf16<2, 4, 4, 2, 0><<<dim3(16, 64), blk, 0, stream>>>(
      hs_bf, D_MODEL, W_in_bf, D_MODEL, xz, 2 * D_INNER, D_MODEL, nullptr, nullptr);

  // 2) xconv = SiLU(conv(x) + b)  (fp32 + bf16)
  conv_silu_kernel<<<(BATCH * SEQ * D_INNER) / 256, blk, 0, stream>>>(
      xz, conv_w, conv_b, xconv, xconv_bf);

  // 3) xdbl = xconv @ W_x^T       M=2048 N=160 K=4096  (fp32 + bf16)
  gemm_wmma_bf16<2, 1, 8, 1, 2><<<dim3(8, 10), blk, 0, stream>>>(
      xconv_bf, D_INNER, W_x_bf, D_INNER, xdbl, XDBL_LD, D_INNER, nullptr, xdbl_bf);

  // 4) delta = softplus(xdbl[:, :128] @ W_dt^T + b_dt)   M=2048 N=4096 K=128
  gemm_wmma_bf16<2, 4, 4, 2, 1><<<dim3(16, 32), blk, 0, stream>>>(
      xdbl_bf, XDBL_LD, W_dt_bf, DT_RANK, delta, D_INNER, DT_RANK, b_dt, nullptr);

  // 5) selective scan (+skip +gate) -> y_bf (reuses xconv_bf storage)
  scan_kernel<<<(BATCH * D_INNER) / 256, blk, 0, stream>>>(
      delta, xconv, xdbl, xz, A_log, D_skip, xconv_bf);

  // 6) out = y @ W_out^T          M=2048 N=2048 K=4096
  gemm_wmma_bf16<2, 4, 4, 2, 0><<<dim3(16, 16), blk, 0, stream>>>(
      xconv_bf, D_INNER, W_out_bf, D_INNER, out, D_MODEL, D_INNER, nullptr, nullptr);
}
